// MultiLinearUpsampling_33260226740563
// MI455X (gfx1250) — matrix operand order repaired and checked by reading, not compiled, not run on hardware
//
#include <hip/hip_runtime.h>
#include <stdint.h>

// ---------------------------------------------------------------------------
// MultiLinearUpsampling on MI455X (gfx1250, wave32, WMMA)
//
// out[b,t,o] = sum_d pooled[b, seg(t), d] * W[l(t), o, d]     (zero if t invalid)
//
// Strategy:
//   * One-time split of pooled and W into bf16 hi/lo pairs (x = hi + lo) in d_ws.
//   * Per workgroup GEMM: D[l][seg, o] = P(32 segs x K) x W[l]^T (K x 16 o),
//     l = 0..15, via v_wmma_f32_16x16x32_bf16 (hi*hi + hi*lo + lo*hi, f32 acc).
//   * Pooled bf16 tiles staged to LDS with the Tensor Data Mover (double
//     buffered, s_wait_tensorcnt); weights streamed from L2 (fits in 192 MB).
//   * Result rows scattered to their t positions (l==15 replicates over tail).
//   * Fallback (small ws): convert-in-kernel variant.
// ---------------------------------------------------------------------------

typedef __attribute__((ext_vector_type(16))) __bf16    v16bf;
typedef __attribute__((ext_vector_type(8)))  __bf16    v8bf;
typedef __attribute__((ext_vector_type(8)))  float     v8f;
typedef __attribute__((ext_vector_type(4)))  unsigned  u32x4;
typedef __attribute__((ext_vector_type(8)))  int       i32x8;
typedef __attribute__((ext_vector_type(4)))  int       i32x4;

#define Bn    8
#define Nn    512
#define DIN   1024
#define DOUT  1024
#define Ln    16
#define Tn    4096

#define KTILE  64                 // d-depth staged per pipeline stage
#define SEGS   32                 // segments (output-tile rows) per workgroup
#define NSTAGE (DIN / KTILE)      // 16 stages

#define NW    ((size_t)Ln * DOUT * DIN)   // 16,777,216 W elements
#define NP    ((size_t)Bn * Nn * DIN)     //  4,194,304 pooled elements

#if __has_builtin(__builtin_amdgcn_tensor_load_to_lds)
#define USE_TDM 1
#else
#define USE_TDM 0
#endif

__global__ void mlu_zero_kernel(float* __restrict__ p, long long n4) {
  long long i = ((long long)blockIdx.x * blockDim.x + threadIdx.x) * 4;
  if (i < n4) {
    float4 z = make_float4(0.f, 0.f, 0.f, 0.f);
    *(float4*)(p + i) = z;
  }
}

// Split x into bf16 hi + bf16 lo (x ~= hi + lo, ~16 mantissa bits kept).
__global__ void mlu_split_kernel(const float* __restrict__ src,
                                 unsigned short* __restrict__ hi,
                                 unsigned short* __restrict__ lo,
                                 int n) {
  int i = (blockIdx.x * blockDim.x + threadIdx.x) * 4;
  if (i + 3 < n) {
    float4 x = *(const float4*)(src + i);
    ushort4 h, l;
    {
      __bf16 h0 = (__bf16)x.x; l.x = __builtin_bit_cast(unsigned short, (__bf16)(x.x - (float)h0)); h.x = __builtin_bit_cast(unsigned short, h0);
      __bf16 h1 = (__bf16)x.y; l.y = __builtin_bit_cast(unsigned short, (__bf16)(x.y - (float)h1)); h.y = __builtin_bit_cast(unsigned short, h1);
      __bf16 h2 = (__bf16)x.z; l.z = __builtin_bit_cast(unsigned short, (__bf16)(x.z - (float)h2)); h.z = __builtin_bit_cast(unsigned short, h2);
      __bf16 h3 = (__bf16)x.w; l.w = __builtin_bit_cast(unsigned short, (__bf16)(x.w - (float)h3)); h.w = __builtin_bit_cast(unsigned short, h3);
    }
    *(ushort4*)(hi + i) = h;
    *(ushort4*)(lo + i) = l;
  }
}

#if USE_TDM
// Issue one TDM 2D tile load: SEGS rows x KTILE elements, row stride DIN,
// packed contiguously (row-major) into LDS at lds_off.
// ds_code: 1 = 2-byte elements (bf16), 2 = 4-byte (f32).
// 6-arg toolchain form: (u32x4 g0, i32x8 g1, i32x4 g2, i32x4 g3, i32x8 g4, cpol)
__device__ __forceinline__ void tdm_load_tile(const void* gsrc, unsigned lds_off, int ds_code) {
  uint64_t ga = (uint64_t)(uintptr_t)gsrc;
  u32x4 g0 = {};
  g0[0] = 1u;                                             // count=1, user D#
  g0[1] = lds_off;                                        // LDS byte address
  g0[2] = (unsigned)(ga & 0xFFFFFFFFu);                   // global_addr[31:0]
  g0[3] = (unsigned)((ga >> 32) & 0x01FFFFFFu) | (2u << 30); // addr[56:32] | type=2
  i32x8 g1 = {};
  g1[0] = (ds_code << 16);                                // data_size
  g1[1] = (int)((unsigned)(DIN & 0xFFFF) << 16);          // tensor_dim0 = 1024
  g1[2] = (int)((unsigned)((Bn * Nn) & 0xFFFF) << 16);    // tensor_dim1 = 4096 rows
  g1[3] = (int)((unsigned)KTILE << 16);                   // tile_dim0 = 64
  g1[4] = SEGS;                                           // tile_dim1 = 32
  g1[5] = DIN;                                            // tensor_dim0_stride = 1024
  i32x4 g2 = {};
  i32x4 g3 = {};
  i32x8 g4 = {};
  __builtin_amdgcn_tensor_load_to_lds(g0, g1, g2, g3, g4, 0);
}
#endif

// ---------------------------------------------------------------------------
// Fast path: operands pre-split to bf16 hi/lo; no conversions in the hot loop.
// ---------------------------------------------------------------------------
__global__ __launch_bounds__(256)
void mlu_wmma_fast_kernel(const unsigned short* __restrict__ phi,  // [B,N,DIN] bf16
                          const unsigned short* __restrict__ plo,
                          const unsigned short* __restrict__ whi,  // [L,DOUT,DIN] bf16
                          const unsigned short* __restrict__ wlo,
                          const int*  __restrict__ idx,
                          float*      __restrict__ out) {
  __shared__ __attribute__((aligned(16))) unsigned short shAhi[2][SEGS * KTILE]; // 8 KB
  __shared__ __attribute__((aligned(16))) unsigned short shAlo[2][SEGS * KTILE]; // 8 KB
  __shared__ int s_start[SEGS];
  __shared__ int s_end[SEGS];

  const int tid  = threadIdx.x;
  const int lane = tid & 31;
  const int wv   = tid >> 5;                 // 0..7 -> levels {2wv, 2wv+1}
  const int lrow = lane & 15;
  const int hi16 = lane >> 4;
  const int ob   = blockIdx.x * 16;
  const int seg0 = blockIdx.y * SEGS;
  const int b    = blockIdx.z;

  if (tid < SEGS) {
    int s = seg0 + tid;
    s_end[tid]   = idx[s];
    s_start[tid] = (s > 0) ? (idx[s - 1] + 1) : 0;
  }

  const unsigned short* Phb = phi + ((size_t)(b * Nn + seg0)) * DIN;
  const unsigned short* Plb = plo + ((size_t)(b * Nn + seg0)) * DIN;

  v8f acc[2][2];
  { v8f z = {}; acc[0][0] = z; acc[0][1] = z; acc[1][0] = z; acc[1][1] = z; }

#if USE_TDM
  const unsigned ldsH0 = (unsigned)(uintptr_t)&shAhi[0][0];
  const unsigned ldsH1 = (unsigned)(uintptr_t)&shAhi[1][0];
  const unsigned ldsL0 = (unsigned)(uintptr_t)&shAlo[0][0];
  const unsigned ldsL1 = (unsigned)(uintptr_t)&shAlo[1][0];
  if (wv == 0) {
    tdm_load_tile(Phb, ldsH0, 1);
    tdm_load_tile(Plb, ldsL0, 1);
  }
#endif
  __syncthreads();

  for (int kt = 0; kt < NSTAGE; ++kt) {
    const int cur = kt & 1;
#if USE_TDM
    if (wv == 0) __builtin_amdgcn_s_wait_tensorcnt(0);
    __syncthreads();
    if (wv == 0 && (kt + 1) < NSTAGE) {
      tdm_load_tile(Phb + (size_t)(kt + 1) * KTILE, cur ? ldsH0 : ldsH1, 1);
      tdm_load_tile(Plb + (size_t)(kt + 1) * KTILE, cur ? ldsL0 : ldsL1, 1);
    }
#else
    {
      // cooperative fallback copy: 32 rows x 64 bf16 (hi and lo)
      const unsigned short* sh = Phb + (size_t)kt * KTILE;
      const unsigned short* sl = Plb + (size_t)kt * KTILE;
      int r = tid >> 3;
      int c = (tid & 7) * 8;
      *(ushort4*)(&shAhi[cur][r * KTILE + c])     = *(const ushort4*)(sh + (size_t)r * DIN + c);
      *(ushort4*)(&shAhi[cur][r * KTILE + c + 4]) = *(const ushort4*)(sh + (size_t)r * DIN + c + 4);
      *(ushort4*)(&shAlo[cur][r * KTILE + c])     = *(const ushort4*)(sl + (size_t)r * DIN + c);
      *(ushort4*)(&shAlo[cur][r * KTILE + c + 4]) = *(const ushort4*)(sl + (size_t)r * DIN + c + 4);
    }
    __syncthreads();
#endif

#pragma unroll
    for (int j = 0; j < 2; ++j) {            // two K=32 WMMA substeps per stage
      const int kk = j * 32;

      // A fragments straight from LDS (16-byte vector reads, no conversion).
      // A k order used here (contiguous, same as B): lane<16 -> K 0..15; lane>=16 -> K 16..31.
      v16bf ah[2], al[2];
#pragma unroll
      for (int st = 0; st < 2; ++st) {
        const int roff = (st * 16 + lrow) * KTILE + kk + hi16 * 16;
        v8bf h0 = *(const v8bf*)(&shAhi[cur][roff]);
        v8bf h1 = *(const v8bf*)(&shAhi[cur][roff + 8]);
        v8bf l0 = *(const v8bf*)(&shAlo[cur][roff]);
        v8bf l1 = *(const v8bf*)(&shAlo[cur][roff + 8]);
#pragma unroll
        for (int i = 0; i < 8; ++i) {
          ah[st][i] = h0[i]; ah[st][8 + i] = h1[i];
          al[st][i] = l0[i]; al[st][8 + i] = l1[i];
        }
      }

#pragma unroll
      for (int li = 0; li < 2; ++li) {
        const int l = wv * 2 + li;
        // B fragment: column n = o; lane<16 -> K 0..15, lane>=16 -> K 16..31.
        // 16 contiguous bf16 -> direct 32-byte vector load into operand regs.
        const size_t boff = ((size_t)l * DOUT + (ob + lrow)) * DIN
                          + (size_t)kt * KTILE + kk + hi16 * 16;
        v16bf bh = *(const v16bf*)(whi + boff);
        v16bf bl = *(const v16bf*)(wlo + boff);
#pragma unroll
        for (int st = 0; st < 2; ++st) {
          acc[li][st] = __builtin_amdgcn_wmma_f32_16x16x32_bf16(
              false, ah[st], false, bh, (short)0, acc[li][st], false, false);
          acc[li][st] = __builtin_amdgcn_wmma_f32_16x16x32_bf16(
              false, ah[st], false, bl, (short)0, acc[li][st], false, false);
          acc[li][st] = __builtin_amdgcn_wmma_f32_16x16x32_bf16(
              false, al[st], false, bh, (short)0, acc[li][st], false, false);
        }
      }
    }
  }

  // Scatter: VGPR v holds (seg = v + 8*hi16 + 16*st, o = ob + lrow).
  const int o = ob + lrow;
  float* outb = out + (size_t)b * Tn * DOUT + o;
#pragma unroll
  for (int li = 0; li < 2; ++li) {
    const int l = wv * 2 + li;
#pragma unroll
    for (int st = 0; st < 2; ++st) {
#pragma unroll
      for (int v = 0; v < 8; ++v) {
        const int sl = st * 16 + v + hi16 * 8;
        const float val = acc[li][st][v];
        const int t0 = s_start[sl] + l;
        const int en = s_end[sl];
        if (l < (Ln - 1)) {
          if (t0 <= en) outb[(size_t)t0 * DOUT] = val;
        } else {
          for (int t = t0; t <= en; ++t) outb[(size_t)t * DOUT] = val;
        }
      }
    }
  }
}

// ---------------------------------------------------------------------------
// Fallback path (ws too small): convert f32 -> bf16 hi/lo inside the kernel.
// ---------------------------------------------------------------------------
__global__ __launch_bounds__(256)
void mlu_wmma_kernel(const float* __restrict__ pooled,
                     const float* __restrict__ Wt,
                     const int*   __restrict__ idx,
                     float*       __restrict__ out) {
  __shared__ float shA[2][SEGS * KTILE];
  __shared__ int s_start[SEGS];
  __shared__ int s_end[SEGS];

  const int tid  = threadIdx.x;
  const int lane = tid & 31;
  const int wv   = tid >> 5;
  const int lrow = lane & 15;
  const int hi16 = lane >> 4;
  const int ob   = blockIdx.x * 16;
  const int seg0 = blockIdx.y * SEGS;
  const int b    = blockIdx.z;

  if (tid < SEGS) {
    int s = seg0 + tid;
    s_end[tid]   = idx[s];
    s_start[tid] = (s > 0) ? (idx[s - 1] + 1) : 0;
  }

  const float* Pbase = pooled + ((size_t)(b * Nn + seg0)) * DIN;

  v8f acc[2][2];
  { v8f z = {}; acc[0][0] = z; acc[0][1] = z; acc[1][0] = z; acc[1][1] = z; }

#if USE_TDM
  const unsigned ldsA0 = (unsigned)(uintptr_t)&shA[0][0];
  const unsigned ldsA1 = (unsigned)(uintptr_t)&shA[1][0];
  if (wv == 0) tdm_load_tile(Pbase, ldsA0, 2);
#endif
  __syncthreads();

  for (int kt = 0; kt < NSTAGE; ++kt) {
    const int cur = kt & 1;
#if USE_TDM
    if (wv == 0) __builtin_amdgcn_s_wait_tensorcnt(0);
    __syncthreads();
    if (wv == 0 && (kt + 1) < NSTAGE)
      tdm_load_tile(Pbase + (size_t)(kt + 1) * KTILE, cur ? ldsA0 : ldsA1, 2);
#else
    {
      const float* src = Pbase + (size_t)kt * KTILE;
      int r = tid >> 3;
      int c = (tid & 7) * 8;
      float4 x0 = *(const float4*)(src + (size_t)r * DIN + c);
      float4 x1 = *(const float4*)(src + (size_t)r * DIN + c + 4);
      *(float4*)(&shA[cur][r * KTILE + c])     = x0;
      *(float4*)(&shA[cur][r * KTILE + c + 4]) = x1;
    }
    __syncthreads();
#endif

#pragma unroll
    for (int j = 0; j < 2; ++j) {
      const int kk = j * 32;
      v16bf ah[2], al[2];
#pragma unroll
      for (int st = 0; st < 2; ++st) {
        const float* ar = &shA[cur][(st * 16 + lrow) * KTILE + kk + hi16 * 16];
#pragma unroll
        for (int i = 0; i < 8; ++i) {
          float x0 = ar[i];
          float x1 = ar[8 + i];
          __bf16 h0 = (__bf16)x0;
          __bf16 h1 = (__bf16)x1;
          ah[st][i]     = h0;  al[st][i]     = (__bf16)(x0 - (float)h0);
          ah[st][8 + i] = h1;  al[st][8 + i] = (__bf16)(x1 - (float)h1);
        }
      }
#pragma unroll
      for (int li = 0; li < 2; ++li) {
        const int l = wv * 2 + li;
        const float* wr = Wt + ((size_t)l * DOUT + (ob + lrow)) * DIN
                             + (size_t)kt * KTILE + kk + hi16 * 16;
        v16bf bh, bl;
#pragma unroll
        for (int i = 0; i < 16; ++i) {
          float x = wr[i];
          __bf16 h = (__bf16)x;
          bh[i] = h;
          bl[i] = (__bf16)(x - (float)h);
        }
#pragma unroll
        for (int st = 0; st < 2; ++st) {
          acc[li][st] = __builtin_amdgcn_wmma_f32_16x16x32_bf16(
              false, ah[st], false, bh, (short)0, acc[li][st], false, false);
          acc[li][st] = __builtin_amdgcn_wmma_f32_16x16x32_bf16(
              false, ah[st], false, bl, (short)0, acc[li][st], false, false);
          acc[li][st] = __builtin_amdgcn_wmma_f32_16x16x32_bf16(
              false, al[st], false, bh, (short)0, acc[li][st], false, false);
        }
      }
    }
  }

  const int o = ob + lrow;
  float* outb = out + (size_t)b * Tn * DOUT + o;
#pragma unroll
  for (int li = 0; li < 2; ++li) {
    const int l = wv * 2 + li;
#pragma unroll
    for (int st = 0; st < 2; ++st) {
#pragma unroll
      for (int v = 0; v < 8; ++v) {
        const int sl = st * 16 + v + hi16 * 8;
        const float val = acc[li][st][v];
        const int t0 = s_start[sl] + l;
        const int en = s_end[sl];
        if (l < (Ln - 1)) {
          if (t0 <= en) outb[(size_t)t0 * DOUT] = val;
        } else {
          for (int t = t0; t <= en; ++t) outb[(size_t)t * DOUT] = val;
        }
      }
    }
  }
}

extern "C" void kernel_launch(void* const* d_in, const int* in_sizes, int n_in,
                              void* d_out, int out_size, void* d_ws, size_t ws_size,
                              hipStream_t stream) {
  (void)in_sizes; (void)n_in; (void)out_size;

  const float* pooled = (const float*)d_in[0];
  const float* Wt     = (const float*)d_in[1];
  const int*   idx    = (const int*)d_in[2];
  float*       out    = (float*)d_out;

  const long long n = (long long)Bn * Tn * DOUT;           // 33,554,432 floats
  mlu_zero_kernel<<<dim3((unsigned)(n / 4 / 256)), dim3(256), 0, stream>>>(out, n);

  dim3 grid(DOUT / 16, Nn / SEGS, Bn);                     // (64, 16, 8)
  const size_t need = (NW + NP) * 2 * sizeof(unsigned short); // ~84 MB

  if (d_ws != nullptr && ws_size >= need) {
    unsigned short* whi = (unsigned short*)d_ws;
    unsigned short* wlo = whi + NW;
    unsigned short* phi = wlo + NW;
    unsigned short* plo = phi + NP;

    mlu_split_kernel<<<dim3((unsigned)(NW / 4 / 256)), dim3(256), 0, stream>>>(
        Wt, whi, wlo, (int)NW);
    mlu_split_kernel<<<dim3((unsigned)(NP / 4 / 256)), dim3(256), 0, stream>>>(
        pooled, phi, plo, (int)NP);
    mlu_wmma_fast_kernel<<<grid, dim3(256), 0, stream>>>(phi, plo, whi, wlo, idx, out);
  } else {
    mlu_wmma_kernel<<<grid, dim3(256), 0, stream>>>(pooled, Wt, idx, out);
  }
}
